// AttnModel3_35012573397062
// MI455X (gfx1250) — matrix operand order repaired and checked by reading, not compiled, not run on hardware
//
#include <hip/hip_runtime.h>
#include <hip/hip_bf16.h>

// ---------------------------------------------------------------------------
// MI455X (gfx1250) fused attention-model kernel.
// 142 GFLOP over ~12MB inputs -> matrix-op bound; use V_WMMA_F32_16X16X32_F16
// (f16 in, f32 accumulate), everything staged in LDS (104KB/block -> 3 blocks
// per 320KB WGP, 8 wave32 per block).
// ---------------------------------------------------------------------------

#define DEVI __device__ __forceinline__

typedef _Float16 half8 __attribute__((ext_vector_type(8)));
typedef _Float16 v16h  __attribute__((ext_vector_type(16)));
typedef float    v8f   __attribute__((ext_vector_type(8)));

// Problem constants
constexpr int BB = 256, NN = 64, FF = 64, SS = 130;
constexpr int SP = 160;                       // S padded to multiple of 32 (5 k-steps)
constexpr float EPS = 1e-6f;
constexpr float INV_SCALE = 0.08770580193070293f;   // 1/sqrt(130)

// LDS layout (bytes)
constexpr int XN_OFF = 0;                     // xn  : 64x160 f16 = 20480
constexpr int Q_OFF  = XN_OFF + FF * SP * 2;  // q   : 64x160 f16 = 20480
constexpr int K_OFF  = Q_OFF  + FF * SP * 2;  // k   : 64x160 f16 = 20480
constexpr int VT_OFF = K_OFF  + FF * SP * 2;  // v^T : 160x64 f16 = 20480
constexpr int P_OFF  = VT_OFF + SP * FF * 2;  // p   : 64x64  f16 =  8192
constexpr int SCR_OFF = P_OFF + FF * FF * 2;  // scratch/scores: 64x64 f32 = 16384
constexpr int SMEM_BYTES = SCR_OFF + FF * FF * 4;   // 106496 bytes total
constexpr int AO_OFF = 0;                     // ao f32 64x160 (40960B) aliases xn+q

// --------------------------- WMMA fragment helpers -------------------------

// A-fragment (16x32, f16) from a row-major f16 LDS matrix.
// lane<16 holds row (lane), K = ks*32 + {0..15};
// lane>=16 holds row (lane-16), K = ks*32 + {16..31}.
DEVI v16h load_a16(const _Float16* base, int stride, int mt, int ks, int lane) {
  const _Float16* p = base + (mt * 16 + (lane & 15)) * stride
                           + ks * 32 + ((lane & 16) ? 16 : 0);
  half8 lo = *(const half8*)(p);
  half8 hi = *(const half8*)(p + 8);
  return __builtin_shufflevector(lo, hi, 0, 1, 2, 3, 4, 5, 6, 7,
                                 8, 9, 10, 11, 12, 13, 14, 15);
}

// B-fragment (32x16, f16) where the source matrix is stored row-major with
// the B *output column* as the row index (i.e. we multiply by its transpose).
// Lane l = column (nt*16 + l%16); 16 contiguous K halves per lane.
DEVI v16h load_b_rows(const _Float16* base, int stride, int nt, int ks, int lane) {
  const _Float16* p = base + (nt * 16 + (lane & 15)) * stride
                           + ks * 32 + ((lane & 16) ? 16 : 0);
  return *(const v16h*)(p);
}

// B-fragment from the prepacked global weight buffer (fragment-major,
// 1KB per fragment, 32 contiguous bytes per lane -> fully coalesced).
DEVI v16h load_b_pack(const _Float16* pack, int fi, int lane) {
  return *(const v16h*)(pack + fi * 512 + lane * 16);
}

DEVI v8f wmma_f16(v16h a, v16h b, v8f c) {
  return __builtin_amdgcn_wmma_f32_16x16x32_f16(
      /*neg_a=*/false, a, /*neg_b=*/false, b,
      /*c_mod=*/(short)0, c, /*reuse_a=*/false, /*reuse_b=*/false);
}

// x[f][j] gather (obs concat + action column), see reference.
DEVI float load_x(const float* __restrict__ h1, const float* __restrict__ h2,
                  const float* __restrict__ sp, int b, int n, int f, int j) {
  if (j < 64)  return h1[(b * 64 + j) * 64 + f];
  if (j < 128) return h2[(b * 64 + (j - 64)) * 64 + f];
  if (j == 128) return sp[b * 64 + f];
  return h2[(b * 64 + n) * 64 + f];   // action column
}

// --------------------------- weight prepack kernel -------------------------
// f32 (130x130) -> zero-padded f16 fragments in WMMA B-operand lane layout.
__global__ __launch_bounds__(256) void prepack_w(
    const float* __restrict__ Wq, const float* __restrict__ Wk,
    const float* __restrict__ Wv, _Float16* __restrict__ pack) {
  int g = blockIdx.x * 256 + threadIdx.x;     // 150 frags * 512 halves = 76800
  if (g >= 150 * 512) return;
  int fi = g >> 9, within = g & 511;
  int lane = within >> 4, h = within & 15;
  int proj = fi / 50, rem = fi % 50, nt = rem / 5, ks = rem % 5;
  int K   = ks * 32 + ((lane & 16) ? 16 : 0) + h;
  int col = nt * 16 + (lane & 15);
  const float* W = (proj == 0) ? Wq : ((proj == 1) ? Wk : Wv);
  float v = (K < SS && col < SS) ? W[K * SS + col] : 0.0f;
  pack[g] = (_Float16)v;
}

// ------------------------------ fused kernel -------------------------------
__global__ __launch_bounds__(256) void attn_fused(
    const float* __restrict__ sp, const float* __restrict__ h1,
    const float* __restrict__ h2,
    const float* __restrict__ bq, const float* __restrict__ bk,
    const float* __restrict__ bv,
    const float* __restrict__ a1, const float* __restrict__ b1,
    const float* __restrict__ a2, const float* __restrict__ b2,
    const float* __restrict__ wlin, const float* __restrict__ blin,
    const _Float16* __restrict__ wpack, float* __restrict__ out) {
  extern __shared__ __align__(16) unsigned char smem[];
  _Float16* xn = (_Float16*)(smem + XN_OFF);
  _Float16* qm = (_Float16*)(smem + Q_OFF);
  _Float16* km = (_Float16*)(smem + K_OFF);
  _Float16* vT = (_Float16*)(smem + VT_OFF);
  _Float16* pm = (_Float16*)(smem + P_OFF);
  float*    scr = (float*)(smem + SCR_OFF);   // 4096 f32: stats scratch / scores
  float*    ao  = (float*)(smem + AO_OFF);    // aliases xn+q (dead by then)

  const int t    = threadIdx.x;
  const int lane = t & 31;
  const int wave = t >> 5;
  const int b = blockIdx.x >> 6;              // batch index
  const int n = blockIdx.x & 63;              // action index

  // ---- Phase 1: norm1 stats (two-pass) + padded f16 xn into LDS ----------
  const int f = t & 63, c = t >> 6;           // 4 threads per feature row
  {
    float s0 = 0.f, s1 = 0.f;
    for (int j = c; j < SS; j += 4) {
      float x = load_x(h1, h2, sp, b, n, f, j);
      s0 += x; s1 += x * x;
    }
    scr[t] = s0; scr[1024 + t] = s1;
  }
  __syncthreads();
  if (t < 64) {
    float su = 0.f, sq = 0.f;
    for (int cc = 0; cc < 4; ++cc) { su += scr[t + cc * 64]; sq += scr[1024 + t + cc * 64]; }
    float m   = su / 130.f;
    float var = sq / 130.f - m * m;
    scr[512 + t] = m;
    scr[576 + t] = 1.f / (sqrtf(fmaxf(var, 0.f)) + EPS);
  }
  __syncthreads();
  {
    float m = scr[512 + f], inv = scr[576 + f];
    for (int j = c; j < SP; j += 4) {
      float v = 0.f;
      if (j < SS) {
        float x = load_x(h1, h2, sp, b, n, f, j);
        v = a1[f * SS + j] * (x - m) * inv + b1[f * SS + j];
      }
      xn[f * SP + j] = (_Float16)v;               // zero pad cols 130..159
    }
  }
  __syncthreads();

  // ---- Phase 2: q/k/v = xn @ W + bias (120 tiles, WMMA) ------------------
  for (int tt = wave; tt < 120; tt += 8) {
    int proj = tt / 40, rem = tt % 40, mt = rem / 10, nt = rem % 10;
    int fbase = (proj * 10 + nt) * 5;
    v8f acc = {};
    for (int ks = 0; ks < 5; ++ks) {
      if (ks < 4)
        __builtin_prefetch(wpack + (fbase + ks + 1) * 512 + lane * 16, 0, 1);
      v16h a  = load_a16(xn, SP, mt, ks, lane);
      v16h bf = load_b_pack(wpack, fbase + ks, lane);
      acc = wmma_f16(a, bf, acc);
    }
    int col = nt * 16 + (lane & 15);
    const float* bias = (proj == 0) ? bq : ((proj == 1) ? bk : bv);
    float badd = (col < SS) ? bias[col] : 0.f;    // pad cols stay exactly 0
    int r0 = (lane & 16) ? 8 : 0;
    if (proj == 2) {                              // store v transposed
      for (int r = 0; r < 8; ++r)
        vT[col * FF + (mt * 16 + r0 + r)] = (_Float16)(acc[r] + badd);
    } else {
      _Float16* dst = (proj == 0) ? qm : km;
      for (int r = 0; r < 8; ++r)
        dst[(mt * 16 + r0 + r) * SP + col] = (_Float16)(acc[r] + badd);
    }
  }
  __syncthreads();

  // ---- Phase 3: sim = q @ k^T / sqrt(S) (16 tiles) -----------------------
  for (int tt = wave * 2; tt < wave * 2 + 2; ++tt) {
    int mt = tt >> 2, nt = tt & 3;
    v8f acc = {};
    for (int ks = 0; ks < 5; ++ks) {
      v16h a  = load_a16(qm, SP, mt, ks, lane);
      v16h bf = load_b_rows(km, SP, nt, ks, lane); // row j of k = B column j
      acc = wmma_f16(a, bf, acc);
    }
    int col = nt * 16 + (lane & 15);
    int r0 = (lane & 16) ? 8 : 0;
    for (int r = 0; r < 8; ++r)
      scr[(mt * 16 + r0 + r) * 64 + col] = acc[r] * INV_SCALE;
  }
  __syncthreads();

  // ---- Phase 4: row softmax -> f16 p -------------------------------------
  if (t < 64) {
    float mx = -3.4e38f;
    for (int j = 0; j < 64; ++j) mx = fmaxf(mx, scr[t * 64 + j]);
    float sum = 0.f;
    for (int j = 0; j < 64; ++j) sum += __expf(scr[t * 64 + j] - mx);
    float rs = 1.f / sum;
    for (int j = 0; j < 64; ++j)
      pm[t * 64 + j] = (_Float16)(__expf(scr[t * 64 + j] - mx) * rs);
  }
  __syncthreads();

  // ---- Phase 5: ao = p @ v, residual x2 (40 tiles, K=64) -----------------
  for (int tt = wave * 5; tt < wave * 5 + 5; ++tt) {
    int mt = tt / 10, nt = tt % 10;
    v8f acc = {};
    for (int ks = 0; ks < 2; ++ks) {
      v16h a  = load_a16(pm, 64, mt, ks, lane);
      v16h bf = load_b_rows(vT, FF, nt, ks, lane); // vT row s = B column s
      acc = wmma_f16(a, bf, acc);
    }
    int col = nt * 16 + (lane & 15);
    int r0 = (lane & 16) ? 8 : 0;
    for (int r = 0; r < 8; ++r)
      ao[(mt * 16 + r0 + r) * SP + col] = 2.0f * acc[r];
  }
  __syncthreads();

  // ---- Phase 6: norm2 + Wlin dot + block reduce --------------------------
  if (t < 64) {
    float su = 0.f, sq = 0.f;
    for (int s = 0; s < SS; ++s) { float v = ao[t * SP + s]; su += v; sq += v * v; }
    float m   = su / 130.f;
    float var = sq / 130.f - m * m;
    float inv = 1.f / (sqrtf(fmaxf(var, 0.f)) + EPS);
    float dot = 0.f;
    for (int s = 0; s < SS; ++s) {
      float v = ao[t * SP + s];
      dot += (a2[t * SS + s] * (v - m) * inv + b2[t * SS + s]) * wlin[t * SS + s];
    }
    scr[t] = dot;
  }
  __syncthreads();
  if (t == 0) {
    float tot = blin[0];
    for (int i = 0; i < 64; ++i) tot += scr[i];
    out[b * NN + n] = tot;
  }
}

// ------------------------------- launcher ----------------------------------
extern "C" void kernel_launch(void* const* d_in, const int* in_sizes, int n_in,
                              void* d_out, int out_size, void* d_ws, size_t ws_size,
                              hipStream_t stream) {
  (void)in_sizes; (void)n_in; (void)out_size; (void)ws_size;
  const float* sp   = (const float*)d_in[0];
  const float* h1   = (const float*)d_in[1];
  const float* h2   = (const float*)d_in[2];
  const float* Wq   = (const float*)d_in[3];
  const float* bq   = (const float*)d_in[4];
  const float* Wk   = (const float*)d_in[5];
  const float* bk   = (const float*)d_in[6];
  const float* Wv   = (const float*)d_in[7];
  const float* bv   = (const float*)d_in[8];
  const float* a1   = (const float*)d_in[9];
  const float* b1   = (const float*)d_in[10];
  const float* a2   = (const float*)d_in[11];
  const float* b2   = (const float*)d_in[12];
  const float* wlin = (const float*)d_in[13];
  const float* blin = (const float*)d_in[14];

  _Float16* pack = (_Float16*)d_ws;             // 150 KB of workspace

  prepack_w<<<300, 256, 0, stream>>>(Wq, Wk, Wv, pack);
  attn_fused<<<BB * NN, 256, SMEM_BYTES, stream>>>(
      sp, h1, h2, bq, bk, bv, a1, b1, a2, b2, wlin, blin, pack, (float*)d_out);
}
